// DigitalMapper_V1_4_46703474377144
// MI455X (gfx1250) — matrix operand order repaired and checked by reading, not compiled, not run on hardware
//
#include <hip/hip_runtime.h>
#include <hip/hip_bf16.h>
#include <math.h>

typedef __attribute__((ext_vector_type(16))) _Float16 v16h;
typedef __attribute__((ext_vector_type(8)))  float    v8f;

#define BATCH 4096
#define IN_F  2048
#define OUT_F 2048

// ---------------------------------------------------------------------------
// Kernel 1: row-wise softmax of raw_weight [OUT_F, IN_F] -> f16 w (same layout)
// One 256-thread block per row; each thread handles IN_F/256 = 8 elements.
// ---------------------------------------------------------------------------
__global__ __launch_bounds__(256) void softmax_rows_f16(
    const float* __restrict__ rw, _Float16* __restrict__ w) {
  const int row = blockIdx.x;
  const int tid = threadIdx.x;
  const float* src = rw + (size_t)row * IN_F;
  _Float16*    dst = w  + (size_t)row * IN_F;

  __shared__ float redmax[8];
  __shared__ float redsum[8];

  float v[8];
#pragma unroll
  for (int j = 0; j < 8; ++j) v[j] = src[tid + j * 256];

  // --- max reduction ---
  float m = v[0];
#pragma unroll
  for (int j = 1; j < 8; ++j) m = fmaxf(m, v[j]);
#pragma unroll
  for (int off = 16; off > 0; off >>= 1) m = fmaxf(m, __shfl_xor(m, off, 32));
  if ((tid & 31) == 0) redmax[tid >> 5] = m;
  __syncthreads();
  m = redmax[0];
#pragma unroll
  for (int i = 1; i < 8; ++i) m = fmaxf(m, redmax[i]);

  // --- exp + sum reduction ---
  float s = 0.f;
#pragma unroll
  for (int j = 0; j < 8; ++j) { v[j] = __expf(v[j] - m); s += v[j]; }
#pragma unroll
  for (int off = 16; off > 0; off >>= 1) s += __shfl_xor(s, off, 32);
  if ((tid & 31) == 0) redsum[tid >> 5] = s;
  __syncthreads();
  s = 0.f;
#pragma unroll
  for (int i = 0; i < 8; ++i) s += redsum[i];
  const float inv = 1.0f / s;

#pragma unroll
  for (int j = 0; j < 8; ++j) dst[tid + j * 256] = (_Float16)(v[j] * inv);
}

// ---------------------------------------------------------------------------
// A-tile loader: builds a 16x32 f16 A operand for one 16-row tile in contiguous k order (B matches).
// elements 0-7  : K = k + 16*half + e      (VGPR0-3)
// elements 8-15 : K = k + 16*half + 8 + e  (VGPR4-7)
// ---------------------------------------------------------------------------
__device__ __forceinline__ v16h load_a_tile(const float* __restrict__ xrow,
                                            int k, int half) {
  const float4* pa0 = (const float4*)(xrow + k + 16 * half);
  const float4* pa1 = (const float4*)(xrow + k + 8 + 16 * half);
  float4 u0 = pa0[0], u1 = pa0[1];
  float4 u2 = pa1[0], u3 = pa1[1];
  v16h a;
  a[0]  = (_Float16)u0.x; a[1]  = (_Float16)u0.y;
  a[2]  = (_Float16)u0.z; a[3]  = (_Float16)u0.w;
  a[4]  = (_Float16)u1.x; a[5]  = (_Float16)u1.y;
  a[6]  = (_Float16)u1.z; a[7]  = (_Float16)u1.w;
  a[8]  = (_Float16)u2.x; a[9]  = (_Float16)u2.y;
  a[10] = (_Float16)u2.z; a[11] = (_Float16)u2.w;
  a[12] = (_Float16)u3.x; a[13] = (_Float16)u3.y;
  a[14] = (_Float16)u3.z; a[15] = (_Float16)u3.w;
  return a;
}

// ---------------------------------------------------------------------------
// Kernel 2: out = ((x @ softmax(w)^T) > 0.5) ? 1 : 0 via v_wmma_f32_16x16x32_f16
//
// Block = 8 waves arranged 4(M) x 2(N); block tile = 128(M) x 128(N).
//   - waves sharing mi read identical A cachelines (WGP$ reuse)
//   - waves sharing ni read identical B cachelines (WGP$ reuse)
// Wave tile = 32(M) x 64(N): 2 A tiles x 4 B tiles = 8 accumulators, so each
// K-step has 8 independent WMMAs to overlap with loads + cvt_pk work.
// Grid = (4096/128) x (2048/128) = 32 x 16 = 512 blocks, exact coverage.
// ---------------------------------------------------------------------------
__global__ __launch_bounds__(256) void gemm_softmaxT_binarize(
    const float* __restrict__ x, const _Float16* __restrict__ w,
    float* __restrict__ out) {
  const int lane = threadIdx.x & 31;
  const int wave = threadIdx.x >> 5;
  const int half = lane >> 4;   // 0: lanes 0-15, 1: lanes 16-31
  const int l16  = lane & 15;

  const int mi = wave & 3;      // 4 M-waves
  const int ni = wave >> 2;     // 2 N-waves
  const int mb = blockIdx.x >> 4;    // 32 M-blocks
  const int nb = blockIdx.x & 15;    // 16 N-blocks
  const int m0 = mb * 128 + mi * 32;
  const int n0 = nb * 128 + ni * 64;

  v8f acc[2][4] = {};

  // A row pointers: tile 0 rows m0..m0+15, tile 1 rows m0+16..m0+31
  const float* xrow0 = x + (size_t)(m0 +      l16) * IN_F;
  const float* xrow1 = x + (size_t)(m0 + 16 + l16) * IN_F;
  // B pointers: lane supplies column n0 + t*16 + l16; row-major w makes each
  // lane's 16 K-values contiguous (32B, aligned).
  const _Float16* wb[4];
#pragma unroll
  for (int t = 0; t < 4; ++t)
    wb[t] = w + (size_t)(n0 + t * 16 + l16) * IN_F + 16 * half;

  for (int k = 0; k < IN_F; k += 32) {
    v16h a0 = load_a_tile(xrow0, k, half);
    v16h a1 = load_a_tile(xrow1, k, half);

    v16h b0 = *(const v16h*)(wb[0] + k);
    v16h b1 = *(const v16h*)(wb[1] + k);
    v16h b2 = *(const v16h*)(wb[2] + k);
    v16h b3 = *(const v16h*)(wb[3] + k);

    acc[0][0] = __builtin_amdgcn_wmma_f32_16x16x32_f16(false, a0, false, b0,
                                                       (short)0, acc[0][0], false, false);
    acc[0][1] = __builtin_amdgcn_wmma_f32_16x16x32_f16(false, a0, false, b1,
                                                       (short)0, acc[0][1], false, false);
    acc[0][2] = __builtin_amdgcn_wmma_f32_16x16x32_f16(false, a0, false, b2,
                                                       (short)0, acc[0][2], false, false);
    acc[0][3] = __builtin_amdgcn_wmma_f32_16x16x32_f16(false, a0, false, b3,
                                                       (short)0, acc[0][3], false, false);
    acc[1][0] = __builtin_amdgcn_wmma_f32_16x16x32_f16(false, a1, false, b0,
                                                       (short)0, acc[1][0], false, false);
    acc[1][1] = __builtin_amdgcn_wmma_f32_16x16x32_f16(false, a1, false, b1,
                                                       (short)0, acc[1][1], false, false);
    acc[1][2] = __builtin_amdgcn_wmma_f32_16x16x32_f16(false, a1, false, b2,
                                                       (short)0, acc[1][2], false, false);
    acc[1][3] = __builtin_amdgcn_wmma_f32_16x16x32_f16(false, a1, false, b3,
                                                       (short)0, acc[1][3], false, false);
  }

  // ---- Epilogue: binarize; C/D layout M = r + 8*half, N = l16 ----
#pragma unroll
  for (int i = 0; i < 2; ++i) {
    float* orow = out + (size_t)(m0 + i * 16 + 8 * half) * OUT_F + n0 + l16;
#pragma unroll
    for (int r = 0; r < 8; ++r) {
      float* p = orow + (size_t)r * OUT_F;
      p[0]  = acc[i][0][r] > 0.5f ? 1.0f : 0.0f;
      p[16] = acc[i][1][r] > 0.5f ? 1.0f : 0.0f;
      p[32] = acc[i][2][r] > 0.5f ? 1.0f : 0.0f;
      p[48] = acc[i][3][r] > 0.5f ? 1.0f : 0.0f;
    }
  }
}

// ---------------------------------------------------------------------------
extern "C" void kernel_launch(void* const* d_in, const int* in_sizes, int n_in,
                              void* d_out, int out_size, void* d_ws, size_t ws_size,
                              hipStream_t stream) {
  const float* x  = (const float*)d_in[0];   // [BATCH, IN_F]
  const float* rw = (const float*)d_in[1];   // [OUT_F, IN_F]
  float* out      = (float*)d_out;           // [BATCH, OUT_F]
  _Float16* w16   = (_Float16*)d_ws;         // 8 MB f16 softmaxed weights

  softmax_rows_f16<<<OUT_F, 256, 0, stream>>>(rw, w16);

  const int blocks = (BATCH / 128) * (OUT_F / 128);   // 512 blocks, 8 waves each
  gemm_softmaxT_binarize<<<blocks, 256, 0, stream>>>(x, w16, out);
}
